// GLSTM_50757923504324
// MI455X (gfx1250) — matrix operand order repaired and checked by reading, not compiled, not run on hardware
//
#include <hip/hip_runtime.h>
#include <stdint.h>
#include <stddef.h>

// ---------------------------------------------------------------------------
// GNN (graph network) pipeline for MI455X / gfx1250.
// All dense GEMMs (D=300, padded to 320) run as bf16 WMMA 16x16x32 with f32
// accumulation. Weights are pre-packed into B-fragment lane order; activations
// are row-major bf16 (stride 320) matching the A-fragment layout, so fragments
// are loaded directly from global memory with b128 loads (no LDS staging).
// Each wave processes TWO 16-row m-tiles so every B fragment feeds 2 WMMAs.
// ---------------------------------------------------------------------------

typedef __attribute__((ext_vector_type(16))) __bf16        v16bf;
typedef __attribute__((ext_vector_type(8)))  float         v8f;
typedef __attribute__((ext_vector_type(4)))  unsigned int  u32x4;

union Frag { v16bf v; u32x4 q[2]; };

__device__ __forceinline__ unsigned short f2bf(float f) {
    unsigned int u = __float_as_uint(f);
    u += 0x7FFFu + ((u >> 16) & 1u);          // round-to-nearest-even
    return (unsigned short)(u >> 16);
}
__device__ __forceinline__ float bf2f(unsigned short h) {
    return __uint_as_float(((unsigned int)h) << 16);
}

#define DIM   300
#define DPAD  320
#define KT    10      // k tiles of 32
#define NTT   20      // n tiles of 16

enum { MODE_PLAIN = 0, MODE_RELU = 1, MODE_EDGE = 2, MODE_RES = 3, MODE_SCATTER = 4 };

// ---------------------------------------------------------------------------
// Generic fused GEMM: C[rows x 320] = A[rows x 320] @ W[320 x 320] (+epilogue)
// 128 threads = 4 waves. Block tile: 32 rows x 320 cols.
//   wave -> group of 5 n-tiles (4 groups * 5 * 16 = 320 cols)
//   each wave handles BOTH 16-row m-tiles (B fragment reused across them)
// rows must be a multiple of 32 (N=100000 and E=400000 both are).
// ---------------------------------------------------------------------------
__global__ __launch_bounds__(128)
void gemm_bf16(const unsigned short* __restrict__ A,
               const unsigned short* __restrict__ Wp,    // packed B fragments
               const float* __restrict__ bias,           // padded [320] or null
               int mode,
               const float* __restrict__ epsArr, int epsIdx,
               const int* __restrict__ ridx,             // edge row index
               const int* __restrict__ cidx,             // edge col index
               const unsigned short* __restrict__ GA,    // gather source (Xin)
               const unsigned short* __restrict__ GB,    // gather source (Xout)
               float* __restrict__ resF,                 // fp32 residual (in-place)
               unsigned short* __restrict__ outB,        // bf16 output
               float* __restrict__ scat)                 // scatter destination
{
    const int tid   = threadIdx.x;
    const int wave  = tid >> 5;             // 0..3 -> n-tiles wave*5 .. +4
    const int lane  = tid & 31;
    const int hi    = lane >> 4;            // K-half selector
    const int lr    = lane & 15;
    const int rowBase = blockIdx.x * 32;

    v8f acc[2][5] = {};

    // A-fragment row pointers for the two m-tiles
    const unsigned short* aRow0 = A + (size_t)(rowBase + lr) * DPAD + hi * 8;
    const unsigned short* aRow1 = aRow0 + (size_t)16 * DPAD;

    #pragma unroll
    for (int kt = 0; kt < KT; ++kt) {
        Frag a0, a1;
        // A-frag (16x32 bf16): lane holds K = base..base+7 and base+16..base+23
        a0.q[0] = *(const u32x4*)(aRow0 + kt * 32);
        a0.q[1] = *(const u32x4*)(aRow0 + kt * 32 + 16);
        a1.q[0] = *(const u32x4*)(aRow1 + kt * 32);
        a1.q[1] = *(const u32x4*)(aRow1 + kt * 32 + 16);
        #pragma unroll
        for (int t = 0; t < 5; ++t) {
            const int ntile = wave * 5 + t;
            const unsigned short* bp =
                Wp + (((size_t)(ntile * KT + kt) * 32 + lane) << 4);
            Frag b;
            b.q[0] = *(const u32x4*)(bp);
            b.q[1] = *(const u32x4*)(bp + 8);
            acc[0][t] = __builtin_amdgcn_wmma_f32_16x16x32_bf16(
                false, a0.v, false, b.v, (short)0, acc[0][t], false, false);
            acc[1][t] = __builtin_amdgcn_wmma_f32_16x16x32_bf16(
                false, a1.v, false, b.v, (short)0, acc[1][t], false, false);
        }
    }

    const float s = epsArr ? (1.0f + epsArr[epsIdx]) : 1.0f;

    #pragma unroll
    for (int m = 0; m < 2; ++m) {
        #pragma unroll
        for (int t = 0; t < 5; ++t) {
            const int ncol = (wave * 5 + t) * 16 + lr;   // C col = lane%16
            const float bv = bias ? bias[ncol] : 0.0f;
            #pragma unroll
            for (int i = 0; i < 8; ++i) {
                const int r = rowBase + m * 16 + i + 8 * hi;  // C row per slot
                float v = acc[m][t][i] + bv;
                if (mode == MODE_EDGE) {
                    const int ra = ridx[r];
                    const int ca = cidx[r];
                    v += bf2f(GA[(size_t)ra * DPAD + ncol]) +
                         bf2f(GB[(size_t)ca * DPAD + ncol]);
                    v = fmaxf(v, 0.0f);
                    outB[(size_t)r * DPAD + ncol] = f2bf(v);
                } else if (mode == MODE_RELU) {
                    v = fmaxf(v, 0.0f);
                    outB[(size_t)r * DPAD + ncol] = f2bf(v);
                } else if (mode == MODE_RES) {
                    const size_t pos = (size_t)r * DPAD + ncol;
                    v += s * resF[pos];
                    resF[pos] = v;
                    outB[pos] = f2bf(v);
                } else if (mode == MODE_SCATTER) {
                    if (ncol < DIM) {
                        const int ca = cidx[r];
                        atomicAdd(&scat[(size_t)ca * DPAD + ncol], v);
                    }
                } else { // MODE_PLAIN
                    outB[(size_t)r * DPAD + ncol] = f2bf(v);
                }
            }
        }
    }
}

// ---------------------------------------------------------------------------
// Pack a 300x300 fp32 weight (row-major, W[k][n]) into per-lane B-fragment
// order, zero-padded to 320x320. blockIdx.x = ntile*10 + ktile, 512 threads.
// Lane L of a (kt,nt) tile holds W[kt*32 + (L>=16)*8 + j%8 + (j>=8)*16][nt*16 + L%16],
// j = 0..15, stored contiguously (16 bf16 = 32 B per lane).
// ---------------------------------------------------------------------------
__global__ void pack_weight(const float* __restrict__ W,
                            unsigned short* __restrict__ Wp)
{
    const int bt   = blockIdx.x;         // 0..199
    const int nt   = bt / KT;
    const int kt   = bt % KT;
    const int t    = threadIdx.x;        // 0..511
    const int lane = t >> 4;
    const int j    = t & 15;
    const int k    = kt * 32 + ((lane >= 16) ? 8 : 0) + (j & 7) + ((j & 8) ? 16 : 0);
    const int n    = nt * 16 + (lane & 15);
    const float v  = (k < DIM && n < DIM) ? W[k * DIM + n] : 0.0f;
    Wp[(size_t)bt * 512 + t] = f2bf(v);
}

__global__ void pack_bias(const float* __restrict__ b, float* __restrict__ bp)
{
    const int j = threadIdx.x;           // 0..319
    bp[j] = (j < DIM) ? b[j] : 0.0f;
}

// Pad a rows x 300 fp32 tensor to rows x 320, producing fp32 master (optional)
// and bf16 shadow. Pad columns are zero (kept zero by all epilogues).
__global__ void pad_convert(const float* __restrict__ src,
                            float* __restrict__ dstF,
                            unsigned short* __restrict__ dstB, int rows)
{
    const int idx = blockIdx.x * blockDim.x + threadIdx.x;
    if (idx >= rows * DPAD) return;
    const int r = idx / DPAD, c = idx - r * DPAD;
    const float v = (c < DIM) ? src[(size_t)r * DIM + c] : 0.0f;
    if (dstF) dstF[idx] = v;
    dstB[idx] = f2bf(v);
}

__global__ void f32_to_bf(const float* __restrict__ src,
                          unsigned short* __restrict__ dst, int n)
{
    const int i = blockIdx.x * blockDim.x + threadIdx.x;
    if (i < n) dst[i] = f2bf(src[i]);
}

__global__ void unpad_out(const float* __restrict__ src,
                          float* __restrict__ dst, int rows)
{
    const int idx = blockIdx.x * blockDim.x + threadIdx.x;
    if (idx >= rows * DIM) return;
    const int r = idx / DIM, c = idx - r * DIM;
    dst[idx] = src[(size_t)r * DPAD + c];
}

// ---------------------------------------------------------------------------
extern "C" void kernel_launch(void* const* d_in, const int* in_sizes, int n_in,
                              void* d_out, int out_size, void* d_ws, size_t ws_size,
                              hipStream_t stream)
{
    const float* x_in    = (const float*)d_in[0];
    const float* e_in    = (const float*)d_in[1];
    const int*   eidx    = (const int*)d_in[2];
    const float* W_edge  = (const float*)d_in[3];
    const float* b_edge  = (const float*)d_in[4];
    const float* W_in    = (const float*)d_in[5];
    const float* W_out   = (const float*)d_in[6];
    const float* W_em1   = (const float*)d_in[7];
    const float* b_em1   = (const float*)d_in[8];
    const float* W_em2   = (const float*)d_in[9];
    const float* b_em2   = (const float*)d_in[10];
    const float* W_n1a   = (const float*)d_in[11];
    const float* b_n1a   = (const float*)d_in[12];
    const float* W_n1b   = (const float*)d_in[13];
    const float* b_n1b   = (const float*)d_in[14];
    const float* W_n2a   = (const float*)d_in[15];
    const float* b_n2a   = (const float*)d_in[16];
    const float* W_n2b   = (const float*)d_in[17];
    const float* b_n2b   = (const float*)d_in[18];
    const float* edge_eps = (const float*)d_in[19];
    const float* node_eps = (const float*)d_in[20];

    const int N = in_sizes[0] / DIM;       // 100000
    const int E = in_sizes[1] / DIM;       // 400000
    const int L = in_sizes[19];            // 3

    const int* ridx = eidx;                // edge_index[0]
    const int* cidx = eidx + E;            // edge_index[1]

    // ---- workspace carving (256B aligned) ----
    uintptr_t p = (uintptr_t)d_ws;
    auto carve = [&](size_t bytes) -> void* {
        uintptr_t r = (p + 255) & ~(uintptr_t)255;
        p = r + bytes;
        return (void*)r;
    };
    float*          ef   = (float*)carve((size_t)E * DPAD * 4);          // edge_attr fp32
    unsigned short* eb   = (unsigned short*)carve((size_t)E * DPAD * 2); // edge_attr bf16
    float*          xf   = (float*)carve((size_t)N * DPAD * 4);          // x fp32
    unsigned short* xb   = (unsigned short*)carve((size_t)N * DPAD * 2); // x bf16
    unsigned short* XinB = (unsigned short*)carve((size_t)N * DPAD * 2);
    unsigned short* XoutB= (unsigned short*)carve((size_t)N * DPAD * 2);
    unsigned short* t1b  = (unsigned short*)carve((size_t)E * DPAD * 2);
    unsigned short* t2b  = (unsigned short*)carve((size_t)E * DPAD * 2);
    float*          agg  = (float*)carve((size_t)N * DPAD * 4);
    unsigned short* aggb = (unsigned short*)carve((size_t)N * DPAD * 2);
    unsigned short* u1b  = (unsigned short*)carve((size_t)N * DPAD * 2);
    const size_t WPM = (size_t)NTT * KT * 512;                           // 102400 ushorts
    unsigned short* Wp = (unsigned short*)carve((size_t)9 * L * WPM * 2);
    float*          Bp = (float*)carve((size_t)7 * L * DPAD * 4);

    // ---- pack all weights / biases (fresh every call: deterministic) ----
    const float* Wsrc[9] = {W_edge, W_in, W_out, W_em1, W_em2, W_n1a, W_n1b, W_n2a, W_n2b};
    const float* bsrc[7] = {b_edge, b_em1, b_em2, b_n1a, b_n1b, b_n2a, b_n2b};
    for (int l = 0; l < L; ++l) {
        for (int m = 0; m < 9; ++m)
            pack_weight<<<NTT * KT, 512, 0, stream>>>(
                Wsrc[m] + (size_t)l * DIM * DIM, Wp + (size_t)(l * 9 + m) * WPM);
        for (int m = 0; m < 7; ++m)
            pack_bias<<<1, DPAD, 0, stream>>>(
                bsrc[m] + (size_t)l * DIM, Bp + (size_t)(l * 7 + m) * DPAD);
    }
    #define WP(l, m) (Wp + (size_t)((l) * 9 + (m)) * WPM)
    #define BB(l, m) (Bp + (size_t)((l) * 7 + (m)) * DPAD)

    // ---- inputs -> padded fp32 master + bf16 shadow ----
    {
        int tn = N * DPAD;
        pad_convert<<<(tn + 255) / 256, 256, 0, stream>>>(x_in, xf, xb, N);
        int te = E * DPAD;
        pad_convert<<<(te + 255) / 256, 256, 0, stream>>>(e_in, ef, eb, E);
    }

    const dim3 blk(128);
    const int gE = E / 32, gN = N / 32;

    for (int l = 0; l < L; ++l) {
        // 1) Xin = x @ W_in ; Xout = x @ W_out   (no bias, no relu)
        gemm_bf16<<<gN, blk, 0, stream>>>(xb, WP(l, 1), nullptr, MODE_PLAIN,
            nullptr, 0, nullptr, nullptr, nullptr, nullptr, nullptr, XinB, nullptr);
        gemm_bf16<<<gN, blk, 0, stream>>>(xb, WP(l, 2), nullptr, MODE_PLAIN,
            nullptr, 0, nullptr, nullptr, nullptr, nullptr, nullptr, XoutB, nullptr);
        // 2) h_e = relu(edge_attr @ W_edge + b + Xin[row] + Xout[col])
        gemm_bf16<<<gE, blk, 0, stream>>>(eb, WP(l, 0), BB(l, 0), MODE_EDGE,
            nullptr, 0, ridx, cidx, XinB, XoutB, nullptr, t1b, nullptr);
        // 3) t = relu(h_e @ W_em1 + b)
        gemm_bf16<<<gE, blk, 0, stream>>>(t1b, WP(l, 3), BB(l, 1), MODE_RELU,
            nullptr, 0, nullptr, nullptr, nullptr, nullptr, nullptr, t2b, nullptr);
        // 4) edge_attr = (1+eps)*edge_attr + (t @ W_em2 + b)   (in-place residual)
        gemm_bf16<<<gE, blk, 0, stream>>>(t2b, WP(l, 4), BB(l, 2), MODE_RES,
            edge_eps, l, nullptr, nullptr, nullptr, nullptr, ef, eb, nullptr);
        // 5) m1 = relu(edge_attr @ W_n1a + b)
        gemm_bf16<<<gE, blk, 0, stream>>>(eb, WP(l, 5), BB(l, 3), MODE_RELU,
            nullptr, 0, nullptr, nullptr, nullptr, nullptr, nullptr, t1b, nullptr);
        // 6) agg = segment_sum(m1 @ W_n1b + b, col)   (fused atomic scatter)
        hipMemsetAsync(agg, 0, (size_t)N * DPAD * 4, stream);
        gemm_bf16<<<gE, blk, 0, stream>>>(t1b, WP(l, 6), BB(l, 4), MODE_SCATTER,
            nullptr, 0, nullptr, cidx, nullptr, nullptr, nullptr, nullptr, agg);
        f32_to_bf<<<(N * DPAD + 255) / 256, 256, 0, stream>>>(agg, aggb, N * DPAD);
        // 7) u1 = relu(agg @ W_n2a + b)
        gemm_bf16<<<gN, blk, 0, stream>>>(aggb, WP(l, 7), BB(l, 5), MODE_RELU,
            nullptr, 0, nullptr, nullptr, nullptr, nullptr, nullptr, u1b, nullptr);
        // 8) x = (1+eps)*x + (u1 @ W_n2b + b)   (in-place residual)
        gemm_bf16<<<gN, blk, 0, stream>>>(u1b, WP(l, 8), BB(l, 6), MODE_RES,
            node_eps, l, nullptr, nullptr, nullptr, nullptr, xf, xb, nullptr);
    }

    // ---- outputs: x then edge_attr, flat fp32 ----
    float* out = (float*)d_out;
    unpad_out<<<(N * DIM + 255) / 256, 256, 0, stream>>>(xf, out, N);
    unpad_out<<<(E * DIM + 255) / 256, 256, 0, stream>>>(ef, out + (size_t)N * DIM, E);
}
